// IGCNet_repara_23536420782217
// MI455X (gfx1250) — matrix operand order repaired and checked by reading, not compiled, not run on hardware
//
#include <hip/hip_runtime.h>
#include <hip/hip_bf16.h>
#include <math.h>

typedef __attribute__((ext_vector_type(16))) _Float16 v16h;
typedef __attribute__((ext_vector_type(8)))  float    v8f;

#define WPB 8           // waves per block (256 threads, wave32)
#define NT2 16

__device__ __forceinline__ void wave_lds_fence() {
  // per-wave ordering of LDS stores vs. loads: DS ops from one wave execute
  // in issue order; just prevent compiler reordering.
  __builtin_amdgcn_wave_barrier();
  asm volatile("" ::: "memory");
}

__device__ __forceinline__ v8f wmma_f16(v16h a, v16h b, v8f c) {
  return __builtin_amdgcn_wmma_f32_16x16x32_f16(
      /*neg_a=*/false, a, /*neg_b=*/false, b,
      /*c_mod=*/(short)0, c, /*reuse_a=*/false, /*reuse_b=*/false);
}

// B fragment (32x16 f16, K-major in memory is W[n][k], row stride = `stride`):
// lane: col n = lane&15, elements h -> k = kbase + h (kbase = ks*32 + hi*16)
__device__ __forceinline__ v16h load_b_frag(const float* __restrict__ W, int stride,
                                            int ncol, int kbase, int kmax) {
  v16h b;
#pragma unroll
  for (int h = 0; h < 16; ++h) {
    int k = kbase + h;
    float v = (k < kmax) ? W[ncol * stride + k] : 0.0f;
    b[h] = (_Float16)v;
  }
  return b;
}

// A fragment (16x32 f16): lane row m = lane&15, hi = lane>>4,
// element h -> feature f = hi*16 + h, read from 32-float row
__device__ __forceinline__ v16h load_a_row32(const float* __restrict__ row, int hi) {
  v16h a;
#pragma unroll
  for (int h = 0; h < 16; ++h) {
    int f = hi * 16 + h;
    a[h] = (_Float16)row[f];
  }
  return a;
}

// A fragment from a 16x64 f16 LDS tile (row-major, stride 64)
__device__ __forceinline__ v16h load_a_lds(const _Float16* __restrict__ hrow,
                                           int m, int ks, int hi) {
  v16h a;
#pragma unroll
  for (int h = 0; h < 16; ++h) {
    int f = ks * 32 + hi * 16 + h;
    a[h] = hrow[m * 64 + f];
  }
  return a;
}

// ---------------- Edge MLP + per-node max aggregation ----------------
// One wave per destination node; its 16 incoming edges are edges 16*node .. 16*node+15.
__global__ __launch_bounds__(256)
void igc_edge_kernel(const float* __restrict__ xin,
                     const int*   __restrict__ src,
                     const float* __restrict__ eattr,
                     const float* __restrict__ w11, const float* __restrict__ b11,
                     const float* __restrict__ w12, const float* __restrict__ b12,
                     float* __restrict__ aggr, int nNodes) {
  __shared__ _Float16 hbuf[WPB][16 * 64];

  const int lane = threadIdx.x & 31;
  const int wid  = threadIdx.x >> 5;
  const int node = blockIdx.x * WPB + wid;
  if (node >= nNodes) return;          // wave-uniform; no block-wide sync used

  const int col = lane & 15;
  const int hi  = lane >> 4;

  // weight fragments, held in VGPRs for the whole kernel
  v16h B1[2][4];                       // w11: [64 out x 40 in], K padded to 64
#pragma unroll
  for (int ks = 0; ks < 2; ++ks)
#pragma unroll
    for (int nt = 0; nt < 4; ++nt)
      B1[ks][nt] = load_b_frag(w11, 40, nt * 16 + col, ks * 32 + hi * 16, 40);
  v16h B2[2][2];                       // w12: [32 out x 64 in]
#pragma unroll
  for (int ks = 0; ks < 2; ++ks)
#pragma unroll
    for (int nt = 0; nt < 2; ++nt)
      B2[ks][nt] = load_b_frag(w12, 64, nt * 16 + col, ks * 32 + hi * 16, 64);

  // A fragments: 16 edges of this node (M = edge-in-node = lane&15)
  const int m = col;
  const int e = node * 16 + m;
  const int s = src[e];
  v16h a0 = load_a_row32(xin + (size_t)s * 32, hi);   // feats 0..31 = x[src]
  v16h a1;                                            // feats 32..39 = edge_attr, rest 0
#pragma unroll
  for (int h = 0; h < 16; ++h) {
    float v = (hi == 0 && h < 8) ? eattr[(size_t)e * 8 + h] : 0.0f;
    a1[h] = (_Float16)v;
  }

  // layer 1: [16x64] = relu([16x64pad] @ w11^T + b11), transposed into LDS
  _Float16* hrow = hbuf[wid];
#pragma unroll
  for (int nt = 0; nt < 4; ++nt) {
    float bv = b11[nt * 16 + col];
    v8f c;
#pragma unroll
    for (int i = 0; i < 8; ++i) c[i] = bv;
    c = wmma_f16(a0, B1[0][nt], c);
    c = wmma_f16(a1, B1[1][nt], c);
#pragma unroll
    for (int r = 0; r < 8; ++r) {
      float v = fmaxf(c[r], 0.0f);
      hrow[(hi * 8 + r) * 64 + nt * 16 + col] = (_Float16)v;
    }
  }
  wave_lds_fence();

  v16h a2k0 = load_a_lds(hrow, m, 0, hi);
  v16h a2k1 = load_a_lds(hrow, m, 1, hi);

  // layer 2: msg = relu([16x64] @ w12^T + b12) -> [16x32]; max over the 16 edges
  float red[2];
#pragma unroll
  for (int nt = 0; nt < 2; ++nt) {
    float bv = b12[nt * 16 + col];
    v8f c;
#pragma unroll
    for (int i = 0; i < 8; ++i) c[i] = bv;
    c = wmma_f16(a2k0, B2[0][nt], c);
    c = wmma_f16(a2k1, B2[1][nt], c);
    float mx = 0.0f;                   // relu => messages >= 0
#pragma unroll
    for (int r = 0; r < 8; ++r) mx = fmaxf(mx, c[r]);
    mx = fmaxf(mx, __shfl_xor(mx, 16, 32));   // combine rows 0-7 with 8-15
    red[nt] = mx;
  }
  // lanes 0-15 carry tile0 cols 0-15, lanes 16-31 carry tile1 cols 0-15
  aggr[(size_t)node * 32 + lane] = (lane < 16) ? red[0] : red[1];
}

// ---------------- Node MLP + norm clip + concat ----------------
// One wave per 16 nodes (M = node-in-tile).
__global__ __launch_bounds__(256)
void igc_node_kernel(const float* __restrict__ xin,
                     const float* __restrict__ aggr,
                     const float* __restrict__ w21, const float* __restrict__ b21,
                     const float* __restrict__ w22, const float* __restrict__ b22,
                     float* __restrict__ out, int nNodes) {
  __shared__ _Float16 hbuf[WPB][16 * 64];
  __shared__ float    cbuf[WPB][16 * 16];

  const int lane = threadIdx.x & 31;
  const int wid  = threadIdx.x >> 5;
  const int wave = blockIdx.x * WPB + wid;
  const int base = wave * 16;
  if (base >= nNodes) return;          // wave-uniform

  const int col = lane & 15;
  const int hi  = lane >> 4;

  v16h B1[2][4];                       // w21: [64 x 64]
#pragma unroll
  for (int ks = 0; ks < 2; ++ks)
#pragma unroll
    for (int nt = 0; nt < 4; ++nt)
      B1[ks][nt] = load_b_frag(w21, 64, nt * 16 + col, ks * 32 + hi * 16, 64);
  v16h B2[2];                          // w22: [16 x 64]
#pragma unroll
  for (int ks = 0; ks < 2; ++ks)
    B2[ks] = load_b_frag(w22, 64, col, ks * 32 + hi * 16, 64);

  const int mnode = base + col;        // this lane's A-row node
  v16h a0 = load_a_row32(xin  + (size_t)mnode * 32, hi);   // feats 0..31 = x
  v16h a1 = load_a_row32(aggr + (size_t)mnode * 32, hi);   // feats 32..63 = aggr

  _Float16* hrow = hbuf[wid];
#pragma unroll
  for (int nt = 0; nt < 4; ++nt) {
    float bv = b21[nt * 16 + col];
    v8f c;
#pragma unroll
    for (int i = 0; i < 8; ++i) c[i] = bv;
    c = wmma_f16(a0, B1[0][nt], c);
    c = wmma_f16(a1, B1[1][nt], c);
#pragma unroll
    for (int r = 0; r < 8; ++r) {
      float v = fmaxf(c[r], 0.0f);
      hrow[(hi * 8 + r) * 64 + nt * 16 + col] = (_Float16)v;
    }
  }
  wave_lds_fence();

  v16h a2k0 = load_a_lds(hrow, col, 0, hi);
  v16h a2k1 = load_a_lds(hrow, col, 1, hi);

  // layer 2: comb = relu([16x64] @ w22^T + b22) -> [16x16]
  float bv = b22[col];
  v8f c;
#pragma unroll
  for (int i = 0; i < 8; ++i) c[i] = bv;
  c = wmma_f16(a2k0, B2[0], c);
  c = wmma_f16(a2k1, B2[1], c);
#pragma unroll
  for (int r = 0; r < 8; ++r) c[r] = fmaxf(c[r], 0.0f);

  // unit-ball clip per row: sum of squares across the 16 columns (lanes in a half)
  v8f ss;
#pragma unroll
  for (int r = 0; r < 8; ++r) ss[r] = c[r] * c[r];
#pragma unroll
  for (int mask = 1; mask < 16; mask <<= 1) {
#pragma unroll
    for (int r = 0; r < 8; ++r) ss[r] += __shfl_xor(ss[r], mask, 32);
  }
#pragma unroll
  for (int r = 0; r < 8; ++r) {
    float nrm = sqrtf(ss[r]);
    float sc  = (nrm > 1.0f) ? (1.0f / nrm) : 1.0f;
    c[r] *= sc;
  }

  // stage comb through LDS, then coalesced write of concat(comb, x[:, :16])
  float* cb = cbuf[wid];
#pragma unroll
  for (int r = 0; r < 8; ++r) cb[(hi * 8 + r) * 16 + col] = c[r];
  wave_lds_fence();

#pragma unroll
  for (int j = 0; j < 16; ++j) {
    int nj = base + j;
    float v = (lane < 16) ? cb[j * 16 + lane]
                          : xin[(size_t)nj * 32 + (lane - 16)];
    out[(size_t)nj * 32 + lane] = v;
  }
}

extern "C" void kernel_launch(void* const* d_in, const int* in_sizes, int n_in,
                              void* d_out, int out_size, void* d_ws, size_t ws_size,
                              hipStream_t stream) {
  const float* x    = (const float*)d_in[0];
  const int*   eidx = (const int*)d_in[1];   // [2, E] int32, row 0 = src
  const float* ea   = (const float*)d_in[2];
  const float* w11  = (const float*)d_in[3];
  const float* b11  = (const float*)d_in[4];
  const float* w12  = (const float*)d_in[5];
  const float* b12  = (const float*)d_in[6];
  const float* w21  = (const float*)d_in[7];
  const float* b21  = (const float*)d_in[8];
  const float* w22  = (const float*)d_in[9];
  const float* b22  = (const float*)d_in[10];

  const int N = in_sizes[0] / 32;
  const int* src = eidx;                     // first E entries

  float* aggr = (float*)d_ws;
  float* hA   = aggr + (size_t)N * 32;
  float* hB   = hA   + (size_t)N * 32;
  float* outF = (float*)d_out;

  const float* cur = x;
  float* nxt[3] = {hA, hB, outF};

  const int eblocks = (N + WPB - 1) / WPB;          // 1 wave per node
  const int nwaves  = (N + 15) / 16;                // 1 wave per 16 nodes
  const int nblocks = (nwaves + WPB - 1) / WPB;

  for (int it = 0; it < 3; ++it) {
    igc_edge_kernel<<<eblocks, 32 * WPB, 0, stream>>>(
        cur, src, ea, w11, b11, w12, b12, aggr, N);
    igc_node_kernel<<<nblocks, 32 * WPB, 0, stream>>>(
        cur, aggr, w21, b21, w22, b22, nxt[it], N);
    cur = nxt[it];
  }
}
